// GINREG_18459769438675
// MI455X (gfx1250) — matrix operand order repaired and checked by reading, not compiled, not run on hardware
//
#include <hip/hip_runtime.h>
#include <stdint.h>

#define N_NODES   50000
#define N_EDGES   800000
#define IN_DIM    128
#define HID       256
#define N_GRAPHS  256
#define LN_EPS    1e-5f

typedef __attribute__((ext_vector_type(16))) __bf16 v16bf;
typedef __attribute__((ext_vector_type(8)))  float  v8f;

union Frag16 { v16bf v; uint4 q[2]; };

__device__ __forceinline__ unsigned short f2bf(float f) {
    union { float f; unsigned int u; } x; x.f = f;
    unsigned int u = x.u;
    return (unsigned short)((u + 0x7FFFu + ((u >> 16) & 1u)) >> 16); // RNE
}
__device__ __forceinline__ float bflo(unsigned w) {
    union { unsigned u; float f; } x; x.u = w << 16; return x.f;
}
__device__ __forceinline__ float bfhi(unsigned w) {
    union { unsigned u; float f; } x; x.u = w & 0xffff0000u; return x.f;
}
__device__ __forceinline__ void atomAdd(float* p, float v) {
    __hip_atomic_fetch_add(p, v, __ATOMIC_RELAXED, __HIP_MEMORY_SCOPE_AGENT);
}

// ---------------------------------------------------------------------------
// Weight prep: W[K x N] fp32 -> Wt[N x K] bf16 (contiguous B fragments)
// ---------------------------------------------------------------------------
__global__ __launch_bounds__(256) void prep_weight(const float* __restrict__ W,
                                                   unsigned short* __restrict__ Wt,
                                                   int K, int N) {
    int gid = blockIdx.x * 256 + threadIdx.x;
    if (gid < K * N) {
        int n = gid / K;
        int k = gid - n * K;
        Wt[gid] = f2bf(W[(size_t)k * N + n]);
    }
}

// fp32 -> bf16, 8 elements per thread (n divisible by 8)
__global__ __launch_bounds__(256) void f32_to_bf16(const float* __restrict__ in,
                                                   unsigned short* __restrict__ out, int n8) {
    int i = blockIdx.x * 256 + threadIdx.x;
    if (i < n8) {
        float4 a = *(const float4*)(in + (size_t)i * 8);
        float4 b = *(const float4*)(in + (size_t)i * 8 + 4);
        uint4 v;
        v.x = (unsigned)f2bf(a.x) | ((unsigned)f2bf(a.y) << 16);
        v.y = (unsigned)f2bf(a.z) | ((unsigned)f2bf(a.w) << 16);
        v.z = (unsigned)f2bf(b.x) | ((unsigned)f2bf(b.y) << 16);
        v.w = (unsigned)f2bf(b.z) | ((unsigned)f2bf(b.w) << 16);
        *(uint4*)(out + (size_t)i * 8) = v;
    }
}

// bf16 -> fp32, 8 elements per thread (agg init: agg = x before edge atomics)
__global__ __launch_bounds__(256) void bf16_to_f32(const unsigned short* __restrict__ in,
                                                   float* __restrict__ out, int n8) {
    int i = blockIdx.x * 256 + threadIdx.x;
    if (i < n8) {
        uint4 v = *(const uint4*)(in + (size_t)i * 8);
        float4 a = { bflo(v.x), bfhi(v.x), bflo(v.y), bfhi(v.y) };
        float4 b = { bflo(v.z), bfhi(v.z), bflo(v.w), bfhi(v.w) };
        *(float4*)(out + (size_t)i * 8) = a;
        *(float4*)(out + (size_t)i * 8 + 4) = b;
    }
}

// ---------------------------------------------------------------------------
// Edge scatter: agg[dst] += x[src]; bf16 gather (16B/thread), fp32 atomics.
// ---------------------------------------------------------------------------
template<int K>
__global__ __launch_bounds__(256) void scatter_edges(const unsigned short* __restrict__ xb,
                                                     const int* __restrict__ src,
                                                     const int* __restrict__ dst,
                                                     float* __restrict__ agg, int nE) {
    const int K8 = K / 8;
    int gid = blockIdx.x * 256 + threadIdx.x;
    int e = gid / K8;
    int c = gid - e * K8;
    if (e < nE) {
        int s = src[e], d = dst[e];
        uint4 v = *(const uint4*)(xb + (size_t)s * K + c * 8);
        float* p = agg + (size_t)d * K + c * 8;
        atomAdd(p + 0, bflo(v.x)); atomAdd(p + 1, bfhi(v.x));
        atomAdd(p + 2, bflo(v.y)); atomAdd(p + 3, bfhi(v.y));
        atomAdd(p + 4, bflo(v.z)); atomAdd(p + 5, bfhi(v.z));
        atomAdd(p + 6, bflo(v.w)); atomAdd(p + 7, bfhi(v.w));
    }
}

// ---------------------------------------------------------------------------
// GEMM1: hb[M x 256] = relu(Af32[M x K] @ W + b)  (bf16 out, sync LDS staging
// with fp32->bf16 conversion). Block tile 64x256, wave tile 16x128.
// ---------------------------------------------------------------------------
template<int K>
__global__ __launch_bounds__(256) void gemm1_kernel(const float* __restrict__ A,
                                                    const unsigned short* __restrict__ Wt,
                                                    const float* __restrict__ bias,
                                                    unsigned short* __restrict__ outb, int M) {
    constexpr int BM = 64;
    constexpr int APAD = 8;
    __shared__ __align__(16) unsigned short sA[BM][K + APAD];

    const int tid  = threadIdx.x;
    const int row0 = blockIdx.x * BM;
    {
        int r  = tid >> 2;                   // 4 threads per row
        int c0 = (tid & 3) * (K / 4);
        int grow = row0 + r; if (grow >= M) grow = M - 1;   // clamp (stores guarded)
        const float* srcp = A + (size_t)grow * K + c0;
        #pragma unroll
        for (int i = 0; i < K / 4; i += 4) {
            float4 v = *(const float4*)(srcp + i);
            sA[r][c0 + i + 0] = f2bf(v.x);
            sA[r][c0 + i + 1] = f2bf(v.y);
            sA[r][c0 + i + 2] = f2bf(v.z);
            sA[r][c0 + i + 3] = f2bf(v.w);
        }
    }
    __syncthreads();

    const int lane = tid & 31, wave = tid >> 5;
    const int wm = wave & 3, wn = wave >> 2;
    const int lm = lane & 15, lh = lane >> 4;

    v8f acc[8] = {};
    for (int k0 = 0; k0 < K; k0 += 32) {
        Frag16 a;
        a.q[0] = *(const uint4*)&sA[wm * 16 + lm][k0 + lh * 8];
        a.q[1] = *(const uint4*)&sA[wm * 16 + lm][k0 + 16 + lh * 8];
        #pragma unroll
        for (int t = 0; t < 8; t++) {
            int n = wn * 128 + t * 16 + lm;
            const unsigned short* wp = Wt + (size_t)n * K + k0 + lh * 8;
            Frag16 b;
            b.q[0] = *(const uint4*)(wp);
            b.q[1] = *(const uint4*)(wp + 16);
            acc[t] = __builtin_amdgcn_wmma_f32_16x16x32_bf16(
                false, a.v, false, b.v, (short)0, acc[t], false, false);
        }
    }
    #pragma unroll
    for (int t = 0; t < 8; t++) {
        int col = wn * 128 + t * 16 + lm;
        float bv = bias[col];
        #pragma unroll
        for (int r = 0; r < 8; r++) {
            int grow = row0 + wm * 16 + lh * 8 + r;       // C layout: M = r + 8*lh
            if (grow < M)
                outb[(size_t)grow * HID + col] = f2bf(fmaxf(acc[t][r] + bv, 0.f));
        }
    }
}

// ---------------------------------------------------------------------------
// GEMM2: out = [LN](relu(Abf16[M x 256] @ W + b))  (bf16 in/out).
// Persistent blocks, double-buffered A tile via GLOBAL_LOAD_ASYNC_TO_LDS_B128:
// the next tile streams into LDS while the current tile's 64 WMMAs execute.
// Every thread issues exactly 8 asyncs per tile (rows clamped, stores guarded),
// so s_wait_asynccnt 8 retires exactly the current tile.
// ---------------------------------------------------------------------------
template<bool DO_LN>
__global__ __launch_bounds__(256) void gemm2_kernel(const unsigned short* __restrict__ Ab,
                                                    const unsigned short* __restrict__ Wt,
                                                    const float* __restrict__ bias,
                                                    const float* __restrict__ gamma,
                                                    const float* __restrict__ beta,
                                                    unsigned short* __restrict__ outb,
                                                    int M, int tiles) {
    constexpr int K = HID;
    constexpr int BM = 64;
    constexpr int APAD = 8;
    __shared__ __align__(16) unsigned short sA[2][BM][K + APAD];
    __shared__ float sSum[BM][2];
    __shared__ float sSq[BM][2];

    const int tid  = threadIdx.x;
    const int lane = tid & 31, wave = tid >> 5;
    const int wm = wave & 3, wn = wave >> 2;
    const int lm = lane & 15, lh = lane >> 4;

    auto issueTile = [&](int row0, int buf) {
        #pragma unroll
        for (int i = 0; i < 8; i++) {            // 8 x 256 chunks = 64 rows x 512B
            int c  = i * 256 + tid;
            int r  = c >> 5;                     // 32 x 16B chunks per row
            int cc = c & 31;
            int grow = row0 + r; if (grow >= M) grow = M - 1;
            unsigned lds = (unsigned)(uintptr_t)&sA[buf][r][cc * 8];
            unsigned long long ga =
                (unsigned long long)(uintptr_t)(Ab + (size_t)grow * K + cc * 8);
            asm volatile("global_load_async_to_lds_b128 %0, %1, off"
                         :: "v"(lds), "v"(ga) : "memory");
        }
    };

    float bv[8], gv[8], bev[8];
    #pragma unroll
    for (int t = 0; t < 8; t++) {
        int col = wn * 128 + t * 16 + lm;
        bv[t] = bias[col];
        gv[t] = DO_LN ? gamma[col] : 0.f;
        bev[t] = DO_LN ? beta[col] : 0.f;
    }

    const int t0 = blockIdx.x, stride = gridDim.x;
    if (t0 < tiles) issueTile(t0 * BM, 0);

    int cb = 0;
    for (int t = t0; t < tiles; t += stride) {
        int nt = t + stride;
        if (nt < tiles) {
            issueTile(nt * BM, cb ^ 1);                      // prefetch next tile
            asm volatile("s_wait_asynccnt 0x8" ::: "memory"); // current tile done
        } else {
            asm volatile("s_wait_asynccnt 0x0" ::: "memory");
        }
        __syncthreads();

        const int row0 = t * BM;
        v8f acc[8] = {};
        for (int k0 = 0; k0 < K; k0 += 32) {
            Frag16 a;
            a.q[0] = *(const uint4*)&sA[cb][wm * 16 + lm][k0 + lh * 8];
            a.q[1] = *(const uint4*)&sA[cb][wm * 16 + lm][k0 + 16 + lh * 8];
            #pragma unroll
            for (int tt = 0; tt < 8; tt++) {
                int n = wn * 128 + tt * 16 + lm;
                const unsigned short* wp = Wt + (size_t)n * K + k0 + lh * 8;
                Frag16 b;
                b.q[0] = *(const uint4*)(wp);
                b.q[1] = *(const uint4*)(wp + 16);
                acc[tt] = __builtin_amdgcn_wmma_f32_16x16x32_bf16(
                    false, a.v, false, b.v, (short)0, acc[tt], false, false);
            }
        }

        if (!DO_LN) {
            #pragma unroll
            for (int tt = 0; tt < 8; tt++) {
                int col = wn * 128 + tt * 16 + lm;
                #pragma unroll
                for (int r = 0; r < 8; r++) {
                    int grow = row0 + wm * 16 + lh * 8 + r;
                    if (grow < M)
                        outb[(size_t)grow * HID + col] = f2bf(fmaxf(acc[tt][r] + bv[tt], 0.f));
                }
            }
        } else {
            #pragma unroll
            for (int r = 0; r < 8; r++) {
                float s = 0.f, s2 = 0.f;
                #pragma unroll
                for (int tt = 0; tt < 8; tt++) {
                    float e = fmaxf(acc[tt][r] + bv[tt], 0.f);
                    s += e; s2 += e * e;
                }
                for (int m = 1; m < 16; m <<= 1) {          // reduce 16-lane half
                    s  += __shfl_xor(s,  m);
                    s2 += __shfl_xor(s2, m);
                }
                int row = wm * 16 + lh * 8 + r;
                if (lm == 0) { sSum[row][wn] = s; sSq[row][wn] = s2; }
            }
            __syncthreads();
            #pragma unroll
            for (int r = 0; r < 8; r++) {
                int row = wm * 16 + lh * 8 + r;
                float s    = sSum[row][0] + sSum[row][1];
                float s2   = sSq[row][0]  + sSq[row][1];
                float mean = s * (1.f / 256.f);
                float rstd = rsqrtf(s2 * (1.f / 256.f) - mean * mean + LN_EPS);
                int grow = row0 + row;
                if (grow < M) {
                    #pragma unroll
                    for (int tt = 0; tt < 8; tt++) {
                        int col = wn * 128 + tt * 16 + lm;
                        float e = fmaxf(acc[tt][r] + bv[tt], 0.f);
                        outb[(size_t)grow * HID + col] =
                            f2bf((e - mean) * rstd * gv[tt] + bev[tt]);
                    }
                }
            }
        }
        __syncthreads();      // all waves done with sA[cb] before it is refilled
        cb ^= 1;
    }
}

// ---------------------------------------------------------------------------
// Mean pool: bf16 gather, fp32 atomic segment sums + counts
// ---------------------------------------------------------------------------
__global__ __launch_bounds__(256) void pool_accum(const unsigned short* __restrict__ xb,
                                                  const int* __restrict__ batch,
                                                  float* __restrict__ sums, int n) {
    int gid = blockIdx.x * 256 + threadIdx.x;
    int node = gid >> 5;                     // 32 chunks of 8 features
    int c = gid & 31;
    if (node < n) {
        int g = batch[node];
        uint4 v = *(const uint4*)(xb + (size_t)node * HID + c * 8);
        float* p = sums + (size_t)g * HID + c * 8;
        atomAdd(p + 0, bflo(v.x)); atomAdd(p + 1, bfhi(v.x));
        atomAdd(p + 2, bflo(v.y)); atomAdd(p + 3, bfhi(v.y));
        atomAdd(p + 4, bflo(v.z)); atomAdd(p + 5, bfhi(v.z));
        atomAdd(p + 6, bflo(v.w)); atomAdd(p + 7, bfhi(v.w));
    }
}

__global__ __launch_bounds__(256) void count_nodes(const int* __restrict__ batch,
                                                   float* __restrict__ counts, int n) {
    int i = blockIdx.x * 256 + threadIdx.x;
    if (i < n) atomAdd(&counts[batch[i]], 1.0f);
}

// ---------------------------------------------------------------------------
// Head: out[g] = relu(pooled @ wp1 + bp1) @ wp2 + bp2 ; one block per graph
// ---------------------------------------------------------------------------
__global__ __launch_bounds__(128) void head_kernel(const float* __restrict__ sums,
                                                   const float* __restrict__ counts,
                                                   const float* __restrict__ wp1,
                                                   const float* __restrict__ bp1,
                                                   const float* __restrict__ wp2,
                                                   const float* __restrict__ bp2,
                                                   float* __restrict__ out) {
    __shared__ float sp[HID];
    __shared__ float part[4];
    int g = blockIdx.x, t = threadIdx.x;
    float cnt = fmaxf(counts[g], 1.0f);
    sp[t]       = sums[(size_t)g * HID + t]       / cnt;
    sp[t + 128] = sums[(size_t)g * HID + t + 128] / cnt;
    __syncthreads();
    float acc = bp1[t];
    #pragma unroll 8
    for (int k = 0; k < HID; k++) acc += sp[k] * wp1[(size_t)k * 128 + t];
    float contrib = fmaxf(acc, 0.f) * wp2[t];
    for (int m = 16; m > 0; m >>= 1) contrib += __shfl_xor(contrib, m);
    if ((t & 31) == 0) part[t >> 5] = contrib;
    __syncthreads();
    if (t == 0) out[g] = part[0] + part[1] + part[2] + part[3] + bp2[0];
}

// ---------------------------------------------------------------------------
extern "C" void kernel_launch(void* const* d_in, const int* in_sizes, int n_in,
                              void* d_out, int out_size, void* d_ws, size_t ws_size,
                              hipStream_t stream) {
    (void)in_sizes; (void)n_in; (void)out_size; (void)ws_size;

    const float* x    = (const float*)d_in[0];
    const int*   eidx = (const int*)d_in[1];
    const int*   batc = (const int*)d_in[2];
    const float* w1_0 = (const float*)d_in[3];
    const float* b1_0 = (const float*)d_in[4];
    const float* w2_0 = (const float*)d_in[5];
    const float* b2_0 = (const float*)d_in[6];
    const float* w1r  = (const float*)d_in[7];
    const float* b1r  = (const float*)d_in[8];
    const float* w2r  = (const float*)d_in[9];
    const float* b2r  = (const float*)d_in[10];
    const float* lng  = (const float*)d_in[11];
    const float* lnb  = (const float*)d_in[12];
    const float* wp1  = (const float*)d_in[13];
    const float* bp1  = (const float*)d_in[14];
    const float* wp2  = (const float*)d_in[15];
    const float* bp2  = (const float*)d_in[16];
    float* out = (float*)d_out;

    char* ws = (char*)d_ws;
    size_t off = 0;
    unsigned short* xb    = (unsigned short*)(ws + off); off += (size_t)N_NODES * HID * 2;
    unsigned short* hb    = (unsigned short*)(ws + off); off += (size_t)N_NODES * HID * 2;
    unsigned short* xb0   = (unsigned short*)(ws + off); off += (size_t)N_NODES * IN_DIM * 2;
    float*          agg   = (float*)(ws + off);          off += (size_t)N_NODES * HID * 4;
    unsigned short* w1_0t = (unsigned short*)(ws + off); off += (size_t)IN_DIM * HID * 2;
    unsigned short* w2_0t = (unsigned short*)(ws + off); off += (size_t)HID * HID * 2;
    unsigned short* w1rt  = (unsigned short*)(ws + off); off += 3 * (size_t)HID * HID * 2;
    unsigned short* w2rt  = (unsigned short*)(ws + off); off += 3 * (size_t)HID * HID * 2;
    float* sums = (float*)(ws + off); off += (size_t)N_GRAPHS * HID * 4;
    float* cnts = (float*)(ws + off); off += (size_t)N_GRAPHS * 4;

    const int* src = eidx;
    const int* dst = eidx + N_EDGES;

    // --- weight prep (transpose + bf16) + bf16 copy of input features ---
    prep_weight<<<(IN_DIM * HID + 255) / 256, 256, 0, stream>>>(w1_0, w1_0t, IN_DIM, HID);
    prep_weight<<<(HID * HID + 255) / 256, 256, 0, stream>>>(w2_0, w2_0t, HID, HID);
    for (int j = 0; j < 3; j++) {
        prep_weight<<<(HID * HID + 255) / 256, 256, 0, stream>>>(
            w1r + (size_t)j * HID * HID, w1rt + (size_t)j * HID * HID, HID, HID);
        prep_weight<<<(HID * HID + 255) / 256, 256, 0, stream>>>(
            w2r + (size_t)j * HID * HID, w2rt + (size_t)j * HID * HID, HID, HID);
    }
    f32_to_bf16<<<(N_NODES * IN_DIM / 8 + 255) / 256, 256, 0, stream>>>(
        x, xb0, N_NODES * IN_DIM / 8);

    const int tiles = (N_NODES + 63) / 64;
    const int PERSIST = 256;

    // --- layer 0 (K = 128) ---
    hipMemcpyAsync(agg, x, (size_t)N_NODES * IN_DIM * 4, hipMemcpyDeviceToDevice, stream);
    scatter_edges<IN_DIM><<<(N_EDGES * (IN_DIM / 8) + 255) / 256, 256, 0, stream>>>(
        xb0, src, dst, agg, N_EDGES);
    gemm1_kernel<IN_DIM><<<tiles, 256, 0, stream>>>(agg, w1_0t, b1_0, hb, N_NODES);
    gemm2_kernel<true><<<PERSIST, 256, 0, stream>>>(
        hb, w2_0t, b2_0, lng, lnb, xb, N_NODES, tiles);

    // --- layers 1..3 (K = 256) ---
    for (int j = 0; j < 3; j++) {
        bf16_to_f32<<<(N_NODES * HID / 8 + 255) / 256, 256, 0, stream>>>(
            xb, agg, N_NODES * HID / 8);
        scatter_edges<HID><<<(N_EDGES * (HID / 8) + 255) / 256, 256, 0, stream>>>(
            xb, src, dst, agg, N_EDGES);
        gemm1_kernel<HID><<<tiles, 256, 0, stream>>>(
            agg, w1rt + (size_t)j * HID * HID, b1r + (size_t)j * HID, hb, N_NODES);
        if (j < 2) {
            gemm2_kernel<true><<<PERSIST, 256, 0, stream>>>(
                hb, w2rt + (size_t)j * HID * HID, b2r + (size_t)j * HID,
                lng + (size_t)(j + 1) * HID, lnb + (size_t)(j + 1) * HID, xb, N_NODES, tiles);
        } else {
            gemm2_kernel<false><<<PERSIST, 256, 0, stream>>>(
                hb, w2rt + (size_t)j * HID * HID, b2r + (size_t)j * HID,
                nullptr, nullptr, xb, N_NODES, tiles);
        }
    }

    // --- mean pool + head ---
    hipMemsetAsync(sums, 0, ((size_t)N_GRAPHS * HID + N_GRAPHS) * 4, stream);
    pool_accum<<<(N_NODES * (HID / 8) + 255) / 256, 256, 0, stream>>>(xb, batc, sums, N_NODES);
    count_nodes<<<(N_NODES + 255) / 256, 256, 0, stream>>>(batc, cnts, N_NODES);
    head_kernel<<<N_GRAPHS, 128, 0, stream>>>(sums, cnts, wp1, bp1, wp2, bp2, out);
}
